// HybridLHM_80582176407603
// MI455X (gfx1250) — compile-verified, matrix operand order checked by reading, not run on hardware
//
#include <hip/hip_runtime.h>
#include <hip/hip_bf16.h>

// ---------------------------------------------------------------------------
// HybridLHM for MI455X (gfx1250).
//  * All dense GEMMs: v_wmma_f32_16x16x32_bf16, 32x32 wave tile (4 wmma / K-step),
//    bf16 operands staged in workspace (weights once per launch, activations
//    once per GEMM) so the inner loop is load_b128 -> wmma with no cvt.
//  * SSM scan: sequential VALU (latency-bound), one thread per (b, channel).
//  * Attention: wave-per-query streaming softmax, LDS score rows.
// ---------------------------------------------------------------------------

#define B_   4
#define L_   1024
#define V_   32000
#define D_   256
#define H_   4
#define DH_  64
#define S_   16
#define DC_  4
#define DI_  512
#define NM_  6
#define NA_  2
#define MBL_ (B_ * L_)

typedef __bf16 bf16;
typedef __attribute__((ext_vector_type(16))) __bf16 v16bf;
typedef __attribute__((ext_vector_type(8)))  __bf16 v8bf;
typedef __attribute__((ext_vector_type(8)))  float  v8f;

__device__ __forceinline__ float sigmoid_f(float x) { return 1.0f / (1.0f + __expf(-x)); }
__device__ __forceinline__ float silu_f(float x)    { return x * sigmoid_f(x); }
__device__ __forceinline__ float softplus_f(float x){ return (x > 20.0f) ? x : log1pf(__expf(x)); }

// ---------------------------------------------------------------------------
// fp32 -> bf16 staging
// ---------------------------------------------------------------------------
__global__ void cvt_bf16_k(const float* __restrict__ src, bf16* __restrict__ dst, int n)
{
    const int i = blockIdx.x * blockDim.x + threadIdx.x;
    if (i < n) dst[i] = (bf16)src[i];
}

// ---------------------------------------------------------------------------
// GEMM: C[M,N] = epilogue( A[M,K] @ W[N,K]^T ), A/W bf16, C fp32.
// EPI bitmask: 1 = +bias[n], 8 = softplus, 4 = silu, 2 = +Res[m,n] (in order).
// 256 threads = 8 waves, wave tile 32x32 (4 accumulators), block tile
// 64(M) x 128(N). M%64==0, N%32==0 (unused waves exit), K%32==0.
//
// bf16 fragment packing per CDNA5 ISA 7.12.2 (16-bit A 16x32):
//   lane groups 0/1 hold K base (lane>>4)*8, elements 0-7 = base..base+7,
//   elements 8-15 = base+16..base+23. B mirrors with column n = lane%16.
// C/D layout: lane L, acc[r] -> (M = r + 8*(L>>4), N = L%16).
// ---------------------------------------------------------------------------
template <int EPI>
__device__ __forceinline__ void store_tile16(
    const v8f& acc, int tm, int tn, int lane,
    const float* __restrict__ bias, const float* __restrict__ Res,
    float* __restrict__ C, int N)
{
    const int n_l = lane & 15;
    const int mh  = lane >> 4;
#pragma unroll
    for (int r = 0; r < 8; ++r) {
        const int row = tm + r + mh * 8;
        const int col = tn + n_l;
        float v = acc[r];
        if (EPI & 1) v += bias[col];
        if (EPI & 8) v = softplus_f(v);
        if (EPI & 4) v = silu_f(v);
        if (EPI & 2) v += Res[(size_t)row * N + col];
        C[(size_t)row * N + col] = v;
    }
}

__device__ __forceinline__ v16bf load_frag(const bf16* __restrict__ p)
{
    const v8bf lo = *(const v8bf*)(p);
    const v8bf hi = *(const v8bf*)(p + 16);
    v16bf f;
#pragma unroll
    for (int i = 0; i < 8; ++i) { f[i] = lo[i]; f[i + 8] = hi[i]; }
    return f;
}

template <int EPI>
__global__ __launch_bounds__(256) void gemm_xwT(
    const bf16* __restrict__ A, const bf16* __restrict__ W,
    const float* __restrict__ bias, const float* __restrict__ Res,
    float* __restrict__ C, int M, int N, int K)
{
    const int lane = threadIdx.x & 31;
    const int wave = threadIdx.x >> 5;
    const int wm   = wave & 1;   // 2 waves along M
    const int wn   = wave >> 1;  // 4 waves along N
    const int tile_m = blockIdx.x * 64  + wm * 32;
    const int tile_n = blockIdx.y * 128 + wn * 32;
    if (tile_m >= M || tile_n >= N) return;

    const int r    = lane & 15;
    const int half = lane >> 4;
    const bf16* a0p = A + (size_t)(tile_m + r) * K      + half * 8;
    const bf16* a1p = a0p + (size_t)16 * K;
    const bf16* b0p = W + (size_t)(tile_n + r) * K      + half * 8;
    const bf16* b1p = b0p + (size_t)16 * K;

    v8f acc00 = {}, acc01 = {}, acc10 = {}, acc11 = {};
#pragma unroll 2
    for (int k0 = 0; k0 < K; k0 += 32) {
        __builtin_prefetch(a0p + k0 + 64, 0, 0);   // gfx1250 global_prefetch_b8
        __builtin_prefetch(b0p + k0 + 64, 0, 0);
        const v16bf a0 = load_frag(a0p + k0);
        const v16bf a1 = load_frag(a1p + k0);
        const v16bf b0 = load_frag(b0p + k0);
        const v16bf b1 = load_frag(b1p + k0);
        acc00 = __builtin_amdgcn_wmma_f32_16x16x32_bf16(false, a0, false, b0, (short)0, acc00, false, false);
        acc01 = __builtin_amdgcn_wmma_f32_16x16x32_bf16(false, a0, false, b1, (short)0, acc01, false, false);
        acc10 = __builtin_amdgcn_wmma_f32_16x16x32_bf16(false, a1, false, b0, (short)0, acc10, false, false);
        acc11 = __builtin_amdgcn_wmma_f32_16x16x32_bf16(false, a1, false, b1, (short)0, acc11, false, false);
    }

    store_tile16<EPI>(acc00, tile_m,      tile_n,      lane, bias, Res, C, N);
    store_tile16<EPI>(acc01, tile_m,      tile_n + 16, lane, bias, Res, C, N);
    store_tile16<EPI>(acc10, tile_m + 16, tile_n,      lane, bias, Res, C, N);
    store_tile16<EPI>(acc11, tile_m + 16, tile_n + 16, lane, bias, Res, C, N);
}

// ---------------------------------------------------------------------------
// LayerNorm: one 256-thread block per row (D_ == 256), biased variance.
// ---------------------------------------------------------------------------
__global__ __launch_bounds__(256) void layernorm_k(
    const float* __restrict__ X, const float* __restrict__ g,
    const float* __restrict__ b, float* __restrict__ Y)
{
    __shared__ float rs[8], rs2[8];
    const int row = blockIdx.x;
    const int tid = threadIdx.x;
    const float x = X[(size_t)row * D_ + tid];
    float s = x, s2 = x * x;
    for (int off = 16; off; off >>= 1) {
        s  += __shfl_xor(s,  off, 32);
        s2 += __shfl_xor(s2, off, 32);
    }
    const int wave = tid >> 5, lane = tid & 31;
    if (lane == 0) { rs[wave] = s; rs2[wave] = s2; }
    __syncthreads();
    if (wave == 0) {
        float t  = (lane < 8) ? rs[lane]  : 0.0f;
        float t2 = (lane < 8) ? rs2[lane] : 0.0f;
        for (int off = 4; off; off >>= 1) {
            t  += __shfl_xor(t,  off, 32);
            t2 += __shfl_xor(t2, off, 32);
        }
        if (lane == 0) { rs[0] = t; rs2[0] = t2; }
    }
    __syncthreads();
    const float mean = rs[0] * (1.0f / D_);
    const float var  = rs2[0] * (1.0f / D_) - mean * mean;
    const float inv  = rsqrtf(var + 1e-5f);
    Y[(size_t)row * D_ + tid] = (x - mean) * inv * g[tid] + b[tid];
}

// ---------------------------------------------------------------------------
// Embedding: base = tok_emb[id] + type_emb[type]; enc = [sin|cos](t*f + ph)
// ---------------------------------------------------------------------------
__global__ void embed_k(
    const int* __restrict__ ids, const int* __restrict__ types,
    const float* __restrict__ ts, const float* __restrict__ tok_emb,
    const float* __restrict__ type_emb, const float* __restrict__ te_freq,
    const float* __restrict__ te_phase, float* __restrict__ base,
    float* __restrict__ enc)
{
    const int idx = blockIdx.x * blockDim.x + threadIdx.x;
    if (idx >= MBL_ * D_) return;
    const int d  = idx & (D_ - 1);
    const int bl = idx / D_;
    base[idx] = tok_emb[(size_t)ids[bl] * D_ + d] + type_emb[(size_t)types[bl] * D_ + d];
    const float t  = ts[bl];
    const int   dh = (d < D_ / 2) ? d : d - D_ / 2;
    const float f  = softplus_f(te_freq[dh]);
    const float ang = t * f + te_phase[dh];
    enc[idx] = (d < D_ / 2) ? sinf(ang) : cosf(ang);
}

// ---------------------------------------------------------------------------
// Depthwise causal conv (DC=4) + bias + silu, reading from xz[:, 0:DI]
// ---------------------------------------------------------------------------
__global__ void convsilu_k(
    const float* __restrict__ xz, const float* __restrict__ conv_w,
    const float* __restrict__ conv_b, float* __restrict__ xb)
{
    const int idx = blockIdx.x * blockDim.x + threadIdx.x;
    if (idx >= MBL_ * DI_) return;
    const int ch = idx & (DI_ - 1);
    const int bl = idx / DI_;
    const int l  = bl & (L_ - 1);
    const int b  = bl / L_;
    float acc = conv_b[ch];
#pragma unroll
    for (int j = 0; j < DC_; ++j) {
        const int lt = l - (DC_ - 1) + j;
        if (lt >= 0)
            acc += conv_w[ch * DC_ + j] * xz[((size_t)(b * L_ + lt)) * (2 * DI_) + ch];
    }
    xb[idx] = silu_f(acc);
}

// ---------------------------------------------------------------------------
// Selective-scan (sequential over L): one thread per (batch, channel).
// Applies the silu(z) gate (z = xz[:, DI:2*DI]).
// ---------------------------------------------------------------------------
__global__ void scan_k(
    const float* __restrict__ xb, const float* __restrict__ bc,
    const float* __restrict__ dt, const float* __restrict__ A_log,
    const float* __restrict__ Dp, const float* __restrict__ xz,
    float* __restrict__ y)
{
    const int c = blockIdx.x * blockDim.x + threadIdx.x;
    if (c >= B_ * DI_) return;
    const int ch = c & (DI_ - 1);
    const int b  = c / DI_;
    float A[S_], h[S_];
#pragma unroll
    for (int s = 0; s < S_; ++s) { A[s] = -__expf(A_log[ch * S_ + s]); h[s] = 0.0f; }
    const float Dpv = Dp[ch];
    for (int t = 0; t < L_; ++t) {
        const size_t bl  = (size_t)b * L_ + t;
        const float  dtv = dt[bl * DI_ + ch];
        const float  xv  = xb[bl * DI_ + ch];
        const float* Bt  = bc + bl * (2 * S_);
        const float* Ct  = Bt + S_;
        const float  dx  = dtv * xv;
        float yv = Dpv * xv;
#pragma unroll
        for (int s = 0; s < S_; ++s) {
            h[s] = __expf(dtv * A[s]) * h[s] + dx * Bt[s];
            yv  += h[s] * Ct[s];
        }
        const float z = xz[bl * (2 * DI_) + DI_ + ch];
        y[bl * DI_ + ch] = yv * silu_f(z);
    }
}

// ---------------------------------------------------------------------------
// Attention: one wave per query, 8 queries per block; scores staged in LDS
// (8 x 1024 f32 = 32 KB of the 320 KB WGP LDS). Causal + mask + time decay.
// ---------------------------------------------------------------------------
__global__ __launch_bounds__(256) void attn_k(
    const float* __restrict__ qkv, const float* __restrict__ ts,
    const int* __restrict__ mask, const float* __restrict__ decay,
    float* __restrict__ out)
{
    __shared__ float sc[8][L_];
    const int wave = threadIdx.x >> 5;
    const int lane = threadIdx.x & 31;
    const int q = blockIdx.x * 8 + wave;
    const int h = blockIdx.y;
    const int b = blockIdx.z;

    const float* qv = qkv + ((size_t)(b * L_ + q) * 3 + 0) * D_ + h * DH_;
    const float  sp = softplus_f(decay[h]);
    const float  tq = ts[b * L_ + q];
    const float  scale = rsqrtf((float)DH_);
    float* row = &sc[wave][0];

    float mx = -3.0e38f;
    for (int k = lane; k <= q; k += 32) {
        const float* kv = qkv + ((size_t)(b * L_ + k) * 3 + 1) * D_ + h * DH_;
        float dot = 0.0f;
#pragma unroll
        for (int d = 0; d < DH_; ++d) dot += qv[d] * kv[d];
        float s = dot * scale - sp * fabsf(tq - ts[b * L_ + k]) * (1.0f / 24.0f);
        if (mask[b * L_ + k] == 0) s = -3.0e38f;
        row[k] = s;
        mx = fmaxf(mx, s);
    }
    for (int off = 16; off; off >>= 1) mx = fmaxf(mx, __shfl_xor(mx, off, 32));
    __syncthreads();

    float sum = 0.0f;
    for (int k = lane; k <= q; k += 32) {
        const float e = __expf(row[k] - mx);
        row[k] = e;
        sum += e;
    }
    for (int off = 16; off; off >>= 1) sum += __shfl_xor(sum, off, 32);
    __syncthreads();
    const float rinv = 1.0f / sum;

#pragma unroll
    for (int dd = 0; dd < 2; ++dd) {
        const int d = lane * 2 + dd;
        float o = 0.0f;
        for (int k = 0; k <= q; ++k) {
            const float* vv = qkv + ((size_t)(b * L_ + k) * 3 + 2) * D_ + h * DH_;
            o += row[k] * vv[d];
        }
        out[((size_t)(b * L_ + q)) * D_ + h * DH_ + d] = o * rinv;
    }
}

// ---------------------------------------------------------------------------
// Scalar heads: readm / mort from cls = xf[:, 0, :]
// ---------------------------------------------------------------------------
__global__ __launch_bounds__(256) void heads_k(
    const float* __restrict__ xf, const float* __restrict__ re_w,
    const float* __restrict__ re_b, const float* __restrict__ mo_w,
    const float* __restrict__ mo_b, float* __restrict__ outp)
{
    __shared__ float r1[256], r2[256];
    const int b = blockIdx.x, tid = threadIdx.x;
    const float* cls = xf + (size_t)b * L_ * D_;
    r1[tid] = cls[tid] * re_w[tid];
    r2[tid] = cls[tid] * mo_w[tid];
    __syncthreads();
    for (int s = 128; s; s >>= 1) {
        if (tid < s) { r1[tid] += r1[tid + s]; r2[tid] += r2[tid + s]; }
        __syncthreads();
    }
    if (tid == 0) {
        outp[b]      = r1[0] + re_b[0];
        outp[B_ + b] = r2[0] + mo_b[0];
    }
}

// ---------------------------------------------------------------------------
// Host orchestration
// ---------------------------------------------------------------------------
static void launch_gemm(int epi, const bf16* A, const bf16* W, const float* bias,
                        const float* Res, float* C, int M, int N, int K,
                        hipStream_t s)
{
    dim3 grid(M / 64, (N + 127) / 128), blk(256);
    switch (epi) {
        case 0: gemm_xwT<0><<<grid, blk, 0, s>>>(A, W, bias, Res, C, M, N, K); break;
        case 2: gemm_xwT<2><<<grid, blk, 0, s>>>(A, W, bias, Res, C, M, N, K); break;
        case 3: gemm_xwT<3><<<grid, blk, 0, s>>>(A, W, bias, Res, C, M, N, K); break;
        case 4: gemm_xwT<4><<<grid, blk, 0, s>>>(A, W, bias, Res, C, M, N, K); break;
        case 9: gemm_xwT<9><<<grid, blk, 0, s>>>(A, W, bias, Res, C, M, N, K); break;
    }
}

static void cvt(const float* src, bf16* dst, size_t n, hipStream_t s)
{
    cvt_bf16_k<<<((int)n + 255) / 256, 256, 0, s>>>(src, dst, (int)n);
}

extern "C" void kernel_launch(void* const* d_in, const int* in_sizes, int n_in,
                              void* d_out, int out_size, void* d_ws, size_t ws_size,
                              hipStream_t stream)
{
    (void)in_sizes; (void)n_in; (void)out_size; (void)ws_size;

    const int*   token_ids   = (const int*)d_in[0];
    const int*   token_types = (const int*)d_in[1];
    const float* timestamps  = (const float*)d_in[2];
    const int*   attn_mask   = (const int*)d_in[3];
    const float* tok_emb  = (const float*)d_in[4];
    const float* type_emb = (const float*)d_in[5];
    const float* te_freq  = (const float*)d_in[6];
    const float* te_phase = (const float*)d_in[7];
    const float* te_w     = (const float*)d_in[8];
    const float* te_b     = (const float*)d_in[9];
    const float* m_norm_g = (const float*)d_in[10];
    const float* m_norm_b = (const float*)d_in[11];
    const float* m_in_w   = (const float*)d_in[12];
    const float* m_conv_w = (const float*)d_in[13];
    const float* m_conv_b = (const float*)d_in[14];
    const float* m_xproj_w= (const float*)d_in[15];
    const float* m_dt_w   = (const float*)d_in[16];
    const float* m_dt_b   = (const float*)d_in[17];
    const float* m_Alog   = (const float*)d_in[18];
    const float* m_D      = (const float*)d_in[19];
    const float* m_out_w  = (const float*)d_in[20];
    const float* a_norm_g = (const float*)d_in[21];
    const float* a_norm_b = (const float*)d_in[22];
    const float* a_qkv_w  = (const float*)d_in[23];
    const float* a_out_w  = (const float*)d_in[24];
    const float* a_decay  = (const float*)d_in[25];
    const float* a_ffg    = (const float*)d_in[26];
    const float* a_ffb    = (const float*)d_in[27];
    const float* a_ff1    = (const float*)d_in[28];
    const float* a_ff2    = (const float*)d_in[29];
    const float* f_g      = (const float*)d_in[30];
    const float* f_b      = (const float*)d_in[31];
    const float* lm_w     = (const float*)d_in[32];
    const float* re_w     = (const float*)d_in[33];
    const float* re_b     = (const float*)d_in[34];
    const float* mo_w     = (const float*)d_in[35];
    const float* mo_b     = (const float*)d_in[36];

    float* out = (float*)d_out;  // [B*L*V lm_logits][B readm][B mort]

    // ---- workspace carve: fp32 activations, then bf16 staging
    float* w = (float*)d_ws;
    float* x    = w; w += (size_t)MBL_ * D_;
    float* xn   = w; w += (size_t)MBL_ * D_;
    float* enc  = w; w += (size_t)MBL_ * D_;
    float* xz   = w; w += (size_t)MBL_ * 2 * DI_;
    float* xb   = w; w += (size_t)MBL_ * DI_;
    float* bc   = w; w += (size_t)MBL_ * 2 * S_;
    float* dt   = w; w += (size_t)MBL_ * DI_;
    float* yb   = w; w += (size_t)MBL_ * DI_;
    float* qkv  = w; w += (size_t)MBL_ * 3 * D_;
    float* ao   = w; w += (size_t)MBL_ * D_;
    float* ffh  = w; w += (size_t)MBL_ * 4 * D_;

    bf16* p = (bf16*)w;
    // bf16 weights (converted once per launch)
    bf16* te_w_bf    = p; p += (size_t)D_ * D_;
    bf16* m_in_w_bf  = p; p += (size_t)NM_ * 2 * DI_ * D_;
    bf16* m_xp_w_bf  = p; p += (size_t)NM_ * 2 * S_ * DI_;
    bf16* m_dt_w_bf  = p; p += (size_t)NM_ * DI_ * DI_;
    bf16* m_out_w_bf = p; p += (size_t)NM_ * D_ * DI_;
    bf16* a_qkv_w_bf = p; p += (size_t)NA_ * 3 * D_ * D_;
    bf16* a_out_w_bf = p; p += (size_t)NA_ * D_ * D_;
    bf16* a_ff1_bf   = p; p += (size_t)NA_ * 4 * D_ * D_;
    bf16* a_ff2_bf   = p; p += (size_t)NA_ * D_ * 4 * D_;
    bf16* lm_w_bf    = p; p += (size_t)V_ * D_;
    // bf16 activation staging (re-used per GEMM, stream-ordered)
    bf16* enc_bf = p; p += (size_t)MBL_ * D_;
    bf16* xn_bf  = p; p += (size_t)MBL_ * D_;
    bf16* xb_bf  = p; p += (size_t)MBL_ * DI_;
    bf16* yb_bf  = p; p += (size_t)MBL_ * DI_;
    bf16* ao_bf  = p; p += (size_t)MBL_ * D_;
    bf16* ffh_bf = p; p += (size_t)MBL_ * 4 * D_;

    // ---- weight staging (bf16, once per launch)
    cvt(te_w,     te_w_bf,    (size_t)D_ * D_, stream);
    cvt(m_in_w,   m_in_w_bf,  (size_t)NM_ * 2 * DI_ * D_, stream);
    cvt(m_xproj_w,m_xp_w_bf,  (size_t)NM_ * 2 * S_ * DI_, stream);
    cvt(m_dt_w,   m_dt_w_bf,  (size_t)NM_ * DI_ * DI_, stream);
    cvt(m_out_w,  m_out_w_bf, (size_t)NM_ * D_ * DI_, stream);
    cvt(a_qkv_w,  a_qkv_w_bf, (size_t)NA_ * 3 * D_ * D_, stream);
    cvt(a_out_w,  a_out_w_bf, (size_t)NA_ * D_ * D_, stream);
    cvt(a_ff1,    a_ff1_bf,   (size_t)NA_ * 4 * D_ * D_, stream);
    cvt(a_ff2,    a_ff2_bf,   (size_t)NA_ * D_ * 4 * D_, stream);
    cvt(lm_w,     lm_w_bf,    (size_t)V_ * D_, stream);

    // ---- embedding: x = tok + type + enc @ te_w^T + te_b
    embed_k<<<(MBL_ * D_ + 255) / 256, 256, 0, stream>>>(
        token_ids, token_types, timestamps, tok_emb, type_emb,
        te_freq, te_phase, x, enc);
    cvt(enc, enc_bf, (size_t)MBL_ * D_, stream);
    launch_gemm(3, enc_bf, te_w_bf, te_b, x, x, MBL_, D_, D_, stream);

    // ---- 8 layers
    int mi = 0, ai = 0;
    for (int i = 0; i < 8; ++i) {
        if ((i + 1) % 4 == 0) {
            // attention block
            layernorm_k<<<MBL_, 256, 0, stream>>>(x, a_norm_g + ai * D_, a_norm_b + ai * D_, xn);
            cvt(xn, xn_bf, (size_t)MBL_ * D_, stream);
            launch_gemm(0, xn_bf, a_qkv_w_bf + (size_t)ai * 3 * D_ * D_, nullptr, nullptr,
                        qkv, MBL_, 3 * D_, D_, stream);
            attn_k<<<dim3(L_ / 8, H_, B_), 256, 0, stream>>>(
                qkv, timestamps, attn_mask, a_decay + ai * H_, ao);
            cvt(ao, ao_bf, (size_t)MBL_ * D_, stream);
            launch_gemm(2, ao_bf, a_out_w_bf + (size_t)ai * D_ * D_, nullptr, x,
                        x, MBL_, D_, D_, stream);
            layernorm_k<<<MBL_, 256, 0, stream>>>(x, a_ffg + ai * D_, a_ffb + ai * D_, xn);
            cvt(xn, xn_bf, (size_t)MBL_ * D_, stream);
            launch_gemm(4, xn_bf, a_ff1_bf + (size_t)ai * 4 * D_ * D_, nullptr, nullptr,
                        ffh, MBL_, 4 * D_, D_, stream);
            cvt(ffh, ffh_bf, (size_t)MBL_ * 4 * D_, stream);
            launch_gemm(2, ffh_bf, a_ff2_bf + (size_t)ai * D_ * 4 * D_, nullptr, x,
                        x, MBL_, D_, 4 * D_, stream);
            ++ai;
        } else {
            // mamba block
            layernorm_k<<<MBL_, 256, 0, stream>>>(x, m_norm_g + mi * D_, m_norm_b + mi * D_, xn);
            cvt(xn, xn_bf, (size_t)MBL_ * D_, stream);
            launch_gemm(0, xn_bf, m_in_w_bf + (size_t)mi * 2 * DI_ * D_, nullptr, nullptr,
                        xz, MBL_, 2 * DI_, D_, stream);
            convsilu_k<<<(MBL_ * DI_ + 255) / 256, 256, 0, stream>>>(
                xz, m_conv_w + (size_t)mi * DI_ * DC_, m_conv_b + mi * DI_, xb);
            cvt(xb, xb_bf, (size_t)MBL_ * DI_, stream);
            launch_gemm(0, xb_bf, m_xp_w_bf + (size_t)mi * 2 * S_ * DI_, nullptr, nullptr,
                        bc, MBL_, 2 * S_, DI_, stream);
            launch_gemm(9, xb_bf, m_dt_w_bf + (size_t)mi * DI_ * DI_, m_dt_b + mi * DI_, nullptr,
                        dt, MBL_, DI_, DI_, stream);
            scan_k<<<(B_ * DI_ + 255) / 256, 256, 0, stream>>>(
                xb, bc, dt, m_Alog + (size_t)mi * DI_ * S_, m_D + mi * DI_, xz, yb);
            cvt(yb, yb_bf, (size_t)MBL_ * DI_, stream);
            launch_gemm(2, yb_bf, m_out_w_bf + (size_t)mi * D_ * DI_, nullptr, x,
                        x, MBL_, D_, DI_, stream);
            ++mi;
        }
    }

    // ---- final LN, LM head (store-bandwidth-bound: 512 MB fp32 logits), heads
    layernorm_k<<<MBL_, 256, 0, stream>>>(x, f_g, f_b, xn);
    cvt(xn, xn_bf, (size_t)MBL_ * D_, stream);
    launch_gemm(0, xn_bf, lm_w_bf, nullptr, nullptr, out, MBL_, V_, D_, stream);
    heads_k<<<B_, 256, 0, stream>>>(xn, re_w, re_b, mo_w, mo_b,
                                    out + (size_t)MBL_ * V_);
}
